// BFEM_72919954751907
// MI455X (gfx1250) — matrix operand order repaired and checked by reading, not compiled, not run on hardware
//
#include <hip/hip_runtime.h>
#include <hip/hip_bf16.h>

typedef __attribute__((ext_vector_type(16))) _Float16 v16h;
typedef __attribute__((ext_vector_type(8)))  _Float16 v8h;
typedef __attribute__((ext_vector_type(8)))  float    v8f;

#define N_   8
#define C_   64
#define HS   64
#define WS_  64
#define HQ   256
#define WQ   256

// workspace byte offsets
#define OFF_WP1  0          // conv1 packed weights: 9*2*4*512 = 36864 f16 (73728 B)
#define OFF_WPF  73728      // flow packed weights:  9*4*512   = 18432 f16 (36864 B)
#define OFF_P4C  110592     // p4c NHWC f16: 8*64*64*64 f16 (4 MB)
#define OFF_FLOW 4304896    // flow: 8*256*256*2 f32 (4 MB)

// ---------------------------------------------------------------------------
// Pack conv weights into WMMA A-fragment order.
// A 16x32 f16 order used here (contiguous k, B the same): lane<16 holds M=lane, K{0..15};
// lane>=16 holds M=lane-16, K{16..31}.
// conv1 fragment id f = (tap*2 + cic)*4 + cot ; flow fragment id f = tap*4 + cic
// flat = f*512 + lane*16 + i
// ---------------------------------------------------------------------------
__global__ void fw_pack_weights(const float* __restrict__ w1,
                                const float* __restrict__ wf,
                                _Float16* __restrict__ wp1,
                                _Float16* __restrict__ wpf) {
    int tid = blockIdx.x * blockDim.x + threadIdx.x;
    if (tid < 36864) {
        int i = tid & 15, lane = (tid >> 4) & 31, f = tid >> 9;
        int cot = f & 3, cic = (f >> 2) & 1, t = f >> 3;
        int lo = (lane < 16) ? 0 : 16;
        int k  = i + lo;
        int co = cot * 16 + (lane & 15);
        int ci = cic * 32 + k;
        wp1[tid] = (_Float16)w1[(co * 64 + ci) * 9 + t];
    } else if (tid < 36864 + 18432) {
        int id = tid - 36864;
        int i = id & 15, lane = (id >> 4) & 31, f = id >> 9;
        int cic = f & 3, t = f >> 2;
        int lo = (lane < 16) ? 0 : 16;
        int k  = i + lo;
        int co = lane & 15;                 // flow has 2 real out-chans, pad to 16
        int ci = cic * 32 + k;
        float v = (co < 2) ? wf[(co * 128 + ci) * 9 + t] : 0.0f;
        wpf[id] = (_Float16)v;
    }
}

// ---------------------------------------------------------------------------
// conv1: 3x3, 64->64, SAME, bias, ReLU.  Implicit GEMM via WMMA.
// Block = 128 threads = 4 waves; wave o owns output-channel tile o (16 ch).
// Output p4c stored NHWC f16.
// ---------------------------------------------------------------------------
__global__ __launch_bounds__(128) void fw_conv1(const float* __restrict__ p4,
                                                const float* __restrict__ bias,
                                                const _Float16* __restrict__ wp1,
                                                _Float16* __restrict__ p4c) {
    __shared__ _Float16 inT[3 * 18 * 64];   // [row][col][ci] f16, 6.9 KB
    const int xb = blockIdx.x * 16;
    const int y  = blockIdx.y;
    const int n  = blockIdx.z;
    const int tid = threadIdx.x;

    for (int idx = tid; idx < 3 * 18 * 64; idx += 128) {
        int col = idx % 18;
        int r   = (idx / 18) % 3;
        int ci  = idx / 54;
        int yy = y - 1 + r, xx = xb - 1 + col;
        float v = 0.0f;
        if (yy >= 0 && yy < HS && xx >= 0 && xx < WS_)
            v = p4[((n * C_ + ci) * HS + yy) * WS_ + xx];
        inT[(r * 18 + col) * 64 + ci] = (_Float16)v;
    }
    __syncthreads();

    const int lane = tid & 31;
    const int o    = tid >> 5;      // output-channel tile
    const int hi   = lane >> 4;
    v8f acc = {};
    for (int t = 0; t < 9; ++t) {
        const int ky = t / 3, kx = t % 3;
        const int col = (lane & 15) + kx;
        #pragma unroll
        for (int cic = 0; cic < 2; ++cic) {
            v16h a = *(const v16h*)(wp1 + (((t * 2 + cic) * 4 + o) * 512 + lane * 16));
            v16h b = *(const v16h*)(inT + ((ky * 18 + col) * 64 + cic * 32 + hi * 16));
            acc = __builtin_amdgcn_wmma_f32_16x16x32_f16(false, a, false, b,
                                                         (short)0, acc, false, false);
        }
    }
    // D layout: VGPR r, lanes 0-15 -> M=r, lanes 16-31 -> M=r+8; N = lane%16
    const int coL = o * 16 + hi * 8;
    const int x   = xb + (lane & 15);
    v8h hv;
    #pragma unroll
    for (int r = 0; r < 8; ++r) {
        float v = acc[r] + bias[coL + r];
        hv[r] = (_Float16)(v > 0.0f ? v : 0.0f);
    }
    *(v8h*)(p4c + (((n * HS + y) * WS_ + x) * C_ + coL)) = hv;
}

// ---------------------------------------------------------------------------
// flow conv: fused (bilinear 4x upsample of p4c) + concat(q1) + 3x3 conv 128->2.
// Block = 128 threads = 4 waves; each wave owns a 16-pixel segment of the
// 64-pixel row tile.  Concat halo staged in LDS as [3][66][128] f16 (50.7 KB).
// ---------------------------------------------------------------------------
__global__ __launch_bounds__(128) void fw_flowconv(const float* __restrict__ q1,
                                                   const _Float16* __restrict__ p4c,
                                                   const _Float16* __restrict__ wpf,
                                                   float* __restrict__ flow) {
    __shared__ _Float16 catT[3 * 66 * 128];
    const int Xb = blockIdx.x * 64;
    const int Y  = blockIdx.y;
    const int n  = blockIdx.z;
    const int tid = threadIdx.x;

    // stage q1 channels [0,64)
    for (int idx = tid; idx < 3 * 66 * 64; idx += 128) {
        int col = idx % 66;
        int r   = (idx / 66) % 3;
        int ci  = idx / 198;
        int yy = Y - 1 + r, xx = Xb - 1 + col;
        float v = 0.0f;
        if (yy >= 0 && yy < HQ && xx >= 0 && xx < WQ)
            v = q1[((n * C_ + ci) * HQ + yy) * WQ + xx];
        catT[(r * 66 + col) * 128 + ci] = (_Float16)v;
    }
    // stage upsampled p4c channels [64,128): half-pixel bilinear, edge clamp
    for (int idx = tid; idx < 3 * 66 * 8; idx += 128) {
        int cg  = idx & 7;
        int col = (idx >> 3) % 66;
        int r   = idx / 528;
        int yy = Y - 1 + r, xx = Xb - 1 + col;
        v8h res;
        if (yy >= 0 && yy < HQ && xx >= 0 && xx < WQ) {
            float sy = yy * 0.25f - 0.375f;
            float sx = xx * 0.25f - 0.375f;
            float fy = floorf(sy), fx = floorf(sx);
            float wy = sy - fy,    wx = sx - fx;
            int y0 = (int)fy, x0 = (int)fx;
            int y0c = min(max(y0, 0), HS - 1),  y1c = min(max(y0 + 1, 0), HS - 1);
            int x0c = min(max(x0, 0), WS_ - 1), x1c = min(max(x0 + 1, 0), WS_ - 1);
            const int cb = cg * 8;
            v8h v00 = *(const v8h*)(p4c + (((n * HS + y0c) * WS_ + x0c) * C_ + cb));
            v8h v01 = *(const v8h*)(p4c + (((n * HS + y0c) * WS_ + x1c) * C_ + cb));
            v8h v10 = *(const v8h*)(p4c + (((n * HS + y1c) * WS_ + x0c) * C_ + cb));
            v8h v11 = *(const v8h*)(p4c + (((n * HS + y1c) * WS_ + x1c) * C_ + cb));
            #pragma unroll
            for (int j = 0; j < 8; ++j) {
                float a0 = (float)v00[j] * (1.f - wx) + (float)v01[j] * wx;
                float a1 = (float)v10[j] * (1.f - wx) + (float)v11[j] * wx;
                res[j] = (_Float16)(a0 * (1.f - wy) + a1 * wy);
            }
        } else {
            #pragma unroll
            for (int j = 0; j < 8; ++j) res[j] = (_Float16)0.0f;
        }
        *(v8h*)(catT + ((r * 66 + col) * 128 + 64 + cg * 8)) = res;
    }
    __syncthreads();

    const int lane = tid & 31;
    const int w    = tid >> 5;      // 16-pixel segment
    const int hi   = lane >> 4;
    v8f acc = {};
    for (int t = 0; t < 9; ++t) {
        const int ky = t / 3, kx = t % 3;
        const int col = w * 16 + (lane & 15) + kx;
        #pragma unroll
        for (int cic = 0; cic < 4; ++cic) {
            v16h a = *(const v16h*)(wpf + ((t * 4 + cic) * 512 + lane * 16));
            v16h b = *(const v16h*)(catT + ((ky * 66 + col) * 128 + cic * 32 + hi * 16));
            acc = __builtin_amdgcn_wmma_f32_16x16x32_f16(false, a, false, b,
                                                         (short)0, acc, false, false);
        }
    }
    if (lane < 16) {                 // M=0 -> acc[0] (flow x), M=1 -> acc[1] (flow y)
        int X = Xb + w * 16 + lane;
        float2 fv = make_float2(acc[0], acc[1]);
        *(float2*)(flow + (((size_t)n * HQ + Y) * WQ + X) * 2) = fv;
    }
}

// ---------------------------------------------------------------------------
// warp: grid = base(-1+2X/255) + flow/256 ; grid_sample bilinear zeros,
// align_corners=False ; out = q1 - warp.  One thread per pixel, 64-ch loop.
// ---------------------------------------------------------------------------
__global__ __launch_bounds__(256) void fw_warp(const float* __restrict__ q1,
                                               const float* __restrict__ flow,
                                               float* __restrict__ out) {
    const int gid = blockIdx.x * 256 + threadIdx.x;
    const int X = gid & 255;
    const int Y = (gid >> 8) & 255;
    const int n = gid >> 16;
    float2 f = *(const float2*)(flow + (size_t)gid * 2);
    float gx = -1.f + (2.f / 255.f) * (float)X + f.x * (1.f / 256.f);
    float gy = -1.f + (2.f / 255.f) * (float)Y + f.y * (1.f / 256.f);
    float ix = ((gx + 1.f) * 256.f - 1.f) * 0.5f;
    float iy = ((gy + 1.f) * 256.f - 1.f) * 0.5f;
    float fx0 = floorf(ix), fy0 = floorf(iy);
    float wx = ix - fx0,    wy = iy - fy0;
    int x0 = (int)fx0, y0 = (int)fy0, x1 = x0 + 1, y1 = y0 + 1;
    float m00 = ((y0 >= 0) & (y0 < HQ) & (x0 >= 0) & (x0 < WQ)) ? 1.f : 0.f;
    float m01 = ((y0 >= 0) & (y0 < HQ) & (x1 >= 0) & (x1 < WQ)) ? 1.f : 0.f;
    float m10 = ((y1 >= 0) & (y1 < HQ) & (x0 >= 0) & (x0 < WQ)) ? 1.f : 0.f;
    float m11 = ((y1 >= 0) & (y1 < HQ) & (x1 >= 0) & (x1 < WQ)) ? 1.f : 0.f;
    float w00 = (1.f - wy) * (1.f - wx) * m00;
    float w01 = (1.f - wy) * wx * m01;
    float w10 = wy * (1.f - wx) * m10;
    float w11 = wy * wx * m11;
    int x0c = min(max(x0, 0), WQ - 1), x1c = min(max(x1, 0), WQ - 1);
    int y0c = min(max(y0, 0), HQ - 1), y1c = min(max(y1, 0), HQ - 1);
    size_t pix = (size_t)Y * WQ + X;
    size_t p00 = (size_t)y0c * WQ + x0c, p01 = (size_t)y0c * WQ + x1c;
    size_t p10 = (size_t)y1c * WQ + x0c, p11 = (size_t)y1c * WQ + x1c;
    const float* qb = q1 + (size_t)n * C_ * HQ * WQ;
    float* ob = out + (size_t)n * C_ * HQ * WQ;
    #pragma unroll 4
    for (int c = 0; c < C_; ++c) {
        const float* qc = qb + (size_t)c * HQ * WQ;
        float s = w00 * qc[p00] + w01 * qc[p01] + w10 * qc[p10] + w11 * qc[p11];
        ob[(size_t)c * HQ * WQ + pix] = qc[pix] - s;
    }
}

extern "C" void kernel_launch(void* const* d_in, const int* in_sizes, int n_in,
                              void* d_out, int out_size, void* d_ws, size_t ws_size,
                              hipStream_t stream) {
    const float* p4      = (const float*)d_in[0];
    const float* q1      = (const float*)d_in[1];
    const float* conv1_w = (const float*)d_in[2];
    const float* conv1_b = (const float*)d_in[3];
    const float* flow_w  = (const float*)d_in[4];
    float* out = (float*)d_out;

    char* ws = (char*)d_ws;
    _Float16* wp1  = (_Float16*)(ws + OFF_WP1);
    _Float16* wpf  = (_Float16*)(ws + OFF_WPF);
    _Float16* p4c  = (_Float16*)(ws + OFF_P4C);
    float*    flow = (float*)(ws + OFF_FLOW);

    fw_pack_weights<<<(36864 + 18432 + 255) / 256, 256, 0, stream>>>(conv1_w, flow_w, wp1, wpf);
    fw_conv1<<<dim3(WS_ / 16, HS, N_), 128, 0, stream>>>(p4, conv1_b, wp1, p4c);
    fw_flowconv<<<dim3(WQ / 64, HQ, N_), 128, 0, stream>>>(q1, p4c, wpf, flow);
    fw_warp<<<(N_ * HQ * WQ) / 256, 256, 0, stream>>>(q1, flow, out);
}
